// CompactedViTAttention_52235392254394
// MI455X (gfx1250) — matrix operand order repaired and checked by reading, not compiled, not run on hardware
//
#include <hip/hip_runtime.h>
#include <hip/hip_bf16.h>

// ---------------------------------------------------------------------------
// CompactedViTAttention for gfx1250 (MI455X): f16 WMMA pipeline.
//   B=16, N=1025 (pad 1088), HIDDEN=768, HEADS=6, HEAD_DIM=64, ALL_HEAD=384
// Round 4: unroll GEMM k-loops x4 so the double-buffer register rotation is
// renamed away instead of materializing v_dual_mov chains each iteration.
// ---------------------------------------------------------------------------

#define BATCH   16
#define NTOK    1025
#define NPAD    1088          // 17 * 64
#define HID     768
#define NHEADS  6
#define DHEAD   64
#define AHEAD   384
#define MROWS   (BATCH * NTOK)  // 16400 flattened token rows
#define LSTR    80              // padded LDS tile row stride (halves), 160B

typedef __attribute__((ext_vector_type(16))) _Float16 v16h;
typedef __attribute__((ext_vector_type(8)))  _Float16 v8h;
typedef __attribute__((ext_vector_type(8)))  float    v8f;
typedef int v4i32 __attribute__((vector_size(16)));

#define AS1 __attribute__((address_space(1)))
#define AS3 __attribute__((address_space(3)))

#if defined(__AMDGCN__) && \
    __has_builtin(__builtin_amdgcn_global_load_async_to_lds_b128) && \
    __has_builtin(__builtin_amdgcn_s_wait_asynccnt)
#define USE_ASYNC_LDS 1
#else
#define USE_ASYNC_LDS 0
#endif

// ----- WMMA fragment loaders (wave32 layouts per CDNA5 ISA 7.12.2) ---------

// A (16x32 f16, MxK): lane l holds row m=l&15; elems 0..7 = K kb..kb+7,
// elems 8..15 = K 16+kb..16+kb+7 with kb=(l>>4)*8.  Row-major source.
__device__ __forceinline__ v16h load_a_frag(const _Float16* base, int lda,
                                            int k, int lane) {
  const int m  = lane & 15;
  const int kb = (lane >> 4) * 8;
  const _Float16* p = base + (size_t)m * lda + k + kb;
  v8h lo = *(const v8h*)(p);
  v8h hi = *(const v8h*)(p + 16);
  v16h a;
#pragma unroll
  for (int i = 0; i < 8; ++i) { a[i] = lo[i]; a[i + 8] = hi[i]; }
  return a;
}

// B (32x16 f16, KxN): lane l holds col n=l&15; elems 0..7 = K kb2..kb2+7,
// elems 8..15 = K 16+kb2..16+kb2+7 with kb2=(l>>4)*8.  Source is TRANSPOSED (bt[n][k], row-major, ldb halves).
__device__ __forceinline__ v16h load_b_frag_t(const _Float16* bt, int ldb,
                                              int n0, int k, int lane) {
  const int n   = n0 + (lane & 15);
  const int kb2 = (lane >> 4) * 8;
  const _Float16* p = bt + (size_t)n * ldb + k + kb2;
  v8h lo = *(const v8h*)(p);
  v8h hi = *(const v8h*)(p + 16);
  v16h b;
#pragma unroll
  for (int i = 0; i < 8; ++i) { b[i] = lo[i]; b[i + 8] = hi[i]; }
  return b;
}

// B fragment from an LDS tile row-major [64][LSTR] (two 8-half runs, A-table K order).
__device__ __forceinline__ v16h load_b_frag_lds(const _Float16* tile,
                                                int row, int koff, int lane) {
  const _Float16* p = tile + (size_t)row * LSTR + koff + ((lane >> 4) * 8);
  v8h lo = *(const v8h*)(p);
  v8h hi = *(const v8h*)(p + 16);
  v16h b;
#pragma unroll
  for (int i = 0; i < 8; ++i) { b[i] = lo[i]; b[i + 8] = hi[i]; }
  return b;
}

__device__ __forceinline__ v8f wmma_f16(v16h a, v16h b, v8f c) {
  return __builtin_amdgcn_wmma_f32_16x16x32_f16(false, a, false, b,
                                                (short)0, c, false, false);
}

// 16B global -> LDS copy: async DMA path (ASYNCcnt) or VGPR fallback.
__device__ __forceinline__ void copy16_g2l(const _Float16* src,
                                           _Float16* dst_lds) {
#if USE_ASYNC_LDS
  __builtin_amdgcn_global_load_async_to_lds_b128(
      (AS1 v4i32*)src, (AS3 v4i32*)dst_lds, 0, 0);
#else
  *(v8h*)dst_lds = *(const v8h*)src;
#endif
}

// ----- K0: fp32 -> f16 transpose-convert for weights -----------------------
__global__ __launch_bounds__(256)
void k_transpose_f16(const float* __restrict__ src, _Float16* __restrict__ dst,
                     int rows, int cols) {
  int idx = blockIdx.x * blockDim.x + threadIdx.x;
  if (idx >= rows * cols) return;
  int r = idx / cols, c = idx % cols;
  dst[(size_t)c * rows + r] = (_Float16)src[idx];
}

// ----- K1: LayerNorm -> f16, padded token rows zeroed ----------------------
__global__ __launch_bounds__(256)
void k_layernorm(const float* __restrict__ x, const float* __restrict__ g,
                 const float* __restrict__ bv, _Float16* __restrict__ out) {
  const int row = blockIdx.x;            // b * NPAD + n
  const int n = row % NPAD;
  _Float16* o = out + (size_t)row * HID;
  if (n >= NTOK) {
    for (int i = threadIdx.x; i < HID; i += 256) o[i] = (_Float16)0.0f;
    return;
  }
  const int b = row / NPAD;
  const float* xr = x + ((size_t)b * NTOK + n) * HID;
  float v[3], s = 0.f, ss = 0.f;
#pragma unroll
  for (int j = 0; j < 3; ++j) {
    v[j] = xr[threadIdx.x + j * 256];
    s += v[j]; ss += v[j] * v[j];
  }
  __shared__ float rs[256], rss[256];
  rs[threadIdx.x] = s; rss[threadIdx.x] = ss;
  __syncthreads();
  for (int off = 128; off > 0; off >>= 1) {
    if (threadIdx.x < off) {
      rs[threadIdx.x]  += rs[threadIdx.x + off];
      rss[threadIdx.x] += rss[threadIdx.x + off];
    }
    __syncthreads();
  }
  const float mu   = rs[0] * (1.0f / HID);
  const float var  = rss[0] * (1.0f / HID) - mu * mu;
  const float rstd = rsqrtf(var + 1e-12f);
#pragma unroll
  for (int j = 0; j < 3; ++j) {
    int c = threadIdx.x + j * 256;
    o[c] = (_Float16)((v[j] - mu) * rstd * g[c] + bv[c]);
  }
}

// ----- K2: QKV GEMM (768 -> 384), WMMA, reg double-buffered ----------------
// grid: (NPAD/64, AHEAD/64, BATCH*3)  block: 128 (4 waves, 16 rows each)
__global__ __launch_bounds__(128)
void k_qkv(const _Float16* __restrict__ normed,
           const _Float16* __restrict__ wqT, const _Float16* __restrict__ wkT,
           const _Float16* __restrict__ wvT,
           const float* __restrict__ bq, const float* __restrict__ bk,
           const float* __restrict__ bv,
           _Float16* __restrict__ Q, _Float16* __restrict__ K,
           _Float16* __restrict__ Vt) {
  const int lane = threadIdx.x & 31;
  const int wave = threadIdx.x >> 5;
  const int bz   = blockIdx.z;
  const int b    = bz / 3;
  const int mat  = bz % 3;
  const int row0 = blockIdx.x * 64 + wave * 16;
  const int col0 = blockIdx.y * 64;
  const _Float16* wT   = (mat == 0) ? wqT : (mat == 1) ? wkT : wvT;
  const float*    bias = (mat == 0) ? bq  : (mat == 1) ? bk  : bv;
  const _Float16* abase = normed + ((size_t)b * NPAD + row0) * HID;

  v8f acc[4] = {v8f{}, v8f{}, v8f{}, v8f{}};
  // software pipeline: fragments for step k live while step k+32 loads issue
  v16h a_cur = load_a_frag(abase, HID, 0, lane);
  v16h b_cur[4];
#pragma unroll
  for (int t = 0; t < 4; ++t)
    b_cur[t] = load_b_frag_t(wT, HID, col0 + t * 16, 0, lane);

#pragma unroll 4
  for (int k = 32; k < HID; k += 32) {
    __builtin_prefetch(abase + ((size_t)(lane & 15)) * HID + k + 32, 0, 1);
    v16h a_nxt = load_a_frag(abase, HID, k, lane);
    v16h b_nxt[4];
#pragma unroll
    for (int t = 0; t < 4; ++t)
      b_nxt[t] = load_b_frag_t(wT, HID, col0 + t * 16, k, lane);
#pragma unroll
    for (int t = 0; t < 4; ++t) acc[t] = wmma_f16(a_cur, b_cur[t], acc[t]);
    a_cur = a_nxt;
#pragma unroll
    for (int t = 0; t < 4; ++t) b_cur[t] = b_nxt[t];
  }
#pragma unroll
  for (int t = 0; t < 4; ++t) acc[t] = wmma_f16(a_cur, b_cur[t], acc[t]);

  const float scale = (mat == 0) ? 0.125f : 1.0f;
#pragma unroll
  for (int t = 0; t < 4; ++t) {
    const int n = col0 + t * 16 + (lane & 15);
    const float bs = bias[n];
    const int h = n / DHEAD, d = n % DHEAD;
#pragma unroll
    for (int r = 0; r < 8; ++r) {
      const int row = row0 + (lane >> 4) * 8 + r;
      const float val = (acc[t][r] + bs) * scale;
      if (mat == 0)
        Q[(((size_t)b * NHEADS + h) * NPAD + row) * DHEAD + d] = (_Float16)val;
      else if (mat == 1)
        K[(((size_t)b * NHEADS + h) * NPAD + row) * DHEAD + d] = (_Float16)val;
      else
        Vt[(((size_t)b * NHEADS + h) * DHEAD + d) * NPAD + row] = (_Float16)val;
    }
  }
}

// ----- K3: flash attention with async double-buffered LDS tiles ------------
// grid: (NPAD/64, NHEADS, BATCH)  block: 128 (4 waves x 16 query rows)
__global__ __launch_bounds__(128)
void k_attn(const _Float16* __restrict__ Q, const _Float16* __restrict__ Km,
            const _Float16* __restrict__ Vt, _Float16* __restrict__ ctx) {
  const int tid  = threadIdx.x;
  const int lane = tid & 31;
  const int wave = tid >> 5;
  const int b = blockIdx.z, h = blockIdx.y;
  const int q0 = blockIdx.x * 64 + wave * 16;
  const _Float16* qbase = Q  + (((size_t)b * NHEADS + h) * NPAD + q0) * DHEAD;
  const _Float16* kbase = Km + ((size_t)b * NHEADS + h) * NPAD * DHEAD;
  const _Float16* vbase = Vt + ((size_t)b * NHEADS + h) * DHEAD * NPAD;

  __shared__ __align__(16) _Float16 kt[2][64][LSTR];  // [key][d]
  __shared__ __align__(16) _Float16 vt[2][64][LSTR];  // [d][key]
  __shared__ _Float16 plds[4][16][72];                // per-wave P relayout

  // Q fragments for the two 32-wide K-dim chunks of head_dim=64.
  v16h aq[2];
  aq[0] = load_a_frag(qbase, DHEAD, 0, lane);
  aq[1] = load_a_frag(qbase, DHEAD, 32, lane);

  // Stage one 64x64 K tile and one 64x64 V tile into LDS (8 b128 per thread).
  auto stage = [&](int buf, int key0) {
#pragma unroll
    for (int i = 0; i < 4; ++i) {
      const int lin = tid + i * 128;
      const int row = lin >> 3, chunk = (lin & 7) * 8;
      copy16_g2l(kbase + (size_t)(key0 + row) * DHEAD + chunk,
                 &kt[buf][row][chunk]);
      copy16_g2l(vbase + (size_t)row * NPAD + key0 + chunk,
                 &vt[buf][row][chunk]);
    }
  };

  v8f acc[4] = {v8f{}, v8f{}, v8f{}, v8f{}};
  float m_i[8], l_i[8];
#pragma unroll
  for (int r = 0; r < 8; ++r) { m_i[r] = -__builtin_inff(); l_i[r] = 0.f; }

  const int NITER = NPAD / 64;  // 17
  stage(0, 0);
  for (int it = 0; it < NITER; ++it) {
    const int buf  = it & 1;
    const int key0 = it * 64;
    if (it + 1 < NITER) {
      stage(buf ^ 1, key0 + 64);      // next tile's DMA overlaps this compute
#if USE_ASYNC_LDS
      __builtin_amdgcn_s_wait_asynccnt(8);   // wait current tile (8 in flight)
#endif
    } else {
#if USE_ASYNC_LDS
      __builtin_amdgcn_s_wait_asynccnt(0);
#endif
    }
    __syncthreads();                  // all waves' copies for tile `it` done

    // ---- S = Q * K^T  (16 x 64 tile, 8 WMMAs; B frags from LDS) ----
    v8f s[4] = {v8f{}, v8f{}, v8f{}, v8f{}};
#pragma unroll
    for (int t = 0; t < 4; ++t) {
      const int keyrow = t * 16 + (lane & 15);
#pragma unroll
      for (int kc = 0; kc < 2; ++kc) {
        v16h bb = load_b_frag_lds(&kt[buf][0][0], keyrow, kc * 32, lane);
        s[t] = wmma_f16(aq[kc], bb, s[t]);
      }
    }
    // ---- mask padded keys ----
#pragma unroll
    for (int t = 0; t < 4; ++t) {
      const int col = key0 + t * 16 + (lane & 15);
      if (col >= NTOK) {
#pragma unroll
        for (int r = 0; r < 8; ++r) s[t][r] = -__builtin_inff();
      }
    }
    // ---- online softmax: row max / exp / row sum (wave32 shfl_xor) ----
    float mnew[8];
#pragma unroll
    for (int r = 0; r < 8; ++r) {
      float mx = s[0][r];
#pragma unroll
      for (int t = 1; t < 4; ++t) mx = fmaxf(mx, s[t][r]);
#pragma unroll
      for (int off = 1; off < 16; off <<= 1)
        mx = fmaxf(mx, __shfl_xor(mx, off, 32));
      mnew[r] = fmaxf(m_i[r], mx);
    }
    float rsum[8];
#pragma unroll
    for (int r = 0; r < 8; ++r) rsum[r] = 0.f;
#pragma unroll
    for (int t = 0; t < 4; ++t)
#pragma unroll
      for (int r = 0; r < 8; ++r) {
        float p = __expf(s[t][r] - mnew[r]);
        s[t][r] = p;
        rsum[r] += p;
      }
#pragma unroll
    for (int r = 0; r < 8; ++r) {
#pragma unroll
      for (int off = 1; off < 16; off <<= 1)
        rsum[r] += __shfl_xor(rsum[r], off, 32);
      const float alpha = __expf(m_i[r] - mnew[r]);
      l_i[r] = l_i[r] * alpha + rsum[r];
      m_i[r] = mnew[r];
#pragma unroll
      for (int t = 0; t < 4; ++t) acc[t][r] *= alpha;
    }
    // ---- relayout P (C layout -> A layout) through LDS ----
#pragma unroll
    for (int t = 0; t < 4; ++t)
#pragma unroll
      for (int r = 0; r < 8; ++r)
        plds[wave][(lane >> 4) * 8 + r][t * 16 + (lane & 15)] =
            (_Float16)s[t][r];
    __syncthreads();
    v16h pa[2];
    {
      const int m = lane & 15, kb = (lane >> 4) * 8;
#pragma unroll
      for (int kc = 0; kc < 2; ++kc) {
        v16h a;
#pragma unroll
        for (int i = 0; i < 8; ++i) {
          a[i]     = plds[wave][m][kc * 32 + kb + i];
          a[i + 8] = plds[wave][m][kc * 32 + 16 + kb + i];
        }
        pa[kc] = a;
      }
    }
    // ---- acc += P * V  (8 WMMAs; V tile is d-major in LDS) ----
#pragma unroll
    for (int t = 0; t < 4; ++t) {
      const int d = t * 16 + (lane & 15);
#pragma unroll
      for (int kc = 0; kc < 2; ++kc) {
        v16h bb = load_b_frag_lds(&vt[buf][0][0], d, kc * 32, lane);
        acc[t] = wmma_f16(pa[kc], bb, acc[t]);
      }
    }
    __syncthreads();   // everyone done with kt/vt[buf] before it is re-staged
  }
  // ---- normalize + store ctx (f16, [B][N][AHEAD]) ----
#pragma unroll
  for (int t = 0; t < 4; ++t) {
    const int d = t * 16 + (lane & 15);
#pragma unroll
    for (int r = 0; r < 8; ++r) {
      const int q = q0 + (lane >> 4) * 8 + r;
      if (q < NTOK) {
        const float val = acc[t][r] / l_i[r];
        ctx[((size_t)b * NTOK + q) * AHEAD + h * DHEAD + d] = (_Float16)val;
      }
    }
  }
}

// ----- K4: output projection (384 -> 768) + bias + residual ----------------
// grid: (ceil(MROWS/64), HID/64)  block: 128
__global__ __launch_bounds__(128)
void k_out(const _Float16* __restrict__ ctx, const _Float16* __restrict__ woT,
           const float* __restrict__ bo, const float* __restrict__ hidden,
           float* __restrict__ out) {
  const int lane = threadIdx.x & 31;
  const int wave = threadIdx.x >> 5;
  const int row0 = blockIdx.x * 64 + wave * 16;
  const int col0 = blockIdx.y * 64;

  const int m = lane & 15;
  int arow = row0 + m;
  if (arow >= MROWS) arow = MROWS - 1;           // clamp; result discarded
  const _Float16* arowp = ctx + (size_t)arow * AHEAD;
  const int kb = (lane >> 4) * 8;

  auto load_a = [&](int k) -> v16h {
    const _Float16* p = arowp + k + kb;
    v8h lo = *(const v8h*)(p);
    v8h hi = *(const v8h*)(p + 16);
    v16h a;
#pragma unroll
    for (int i = 0; i < 8; ++i) { a[i] = lo[i]; a[i + 8] = hi[i]; }
    return a;
  };

  v8f acc[4] = {v8f{}, v8f{}, v8f{}, v8f{}};
  v16h a_cur = load_a(0);
  v16h b_cur[4];
#pragma unroll
  for (int t = 0; t < 4; ++t)
    b_cur[t] = load_b_frag_t(woT, AHEAD, col0 + t * 16, 0, lane);

#pragma unroll 4
  for (int k = 32; k < AHEAD; k += 32) {
    __builtin_prefetch(arowp + k + 32, 0, 1);
    v16h a_nxt = load_a(k);
    v16h b_nxt[4];
#pragma unroll
    for (int t = 0; t < 4; ++t)
      b_nxt[t] = load_b_frag_t(woT, AHEAD, col0 + t * 16, k, lane);
#pragma unroll
    for (int t = 0; t < 4; ++t) acc[t] = wmma_f16(a_cur, b_cur[t], acc[t]);
    a_cur = a_nxt;
#pragma unroll
    for (int t = 0; t < 4; ++t) b_cur[t] = b_nxt[t];
  }
#pragma unroll
  for (int t = 0; t < 4; ++t) acc[t] = wmma_f16(a_cur, b_cur[t], acc[t]);

#pragma unroll
  for (int t = 0; t < 4; ++t) {
    const int c = col0 + t * 16 + (lane & 15);
    const float bias = bo[c];
#pragma unroll
    for (int r = 0; r < 8; ++r) {
      const int row = row0 + (lane >> 4) * 8 + r;
      if (row < MROWS) {
        const size_t idx = (size_t)row * HID + c;
        out[idx] = hidden[idx] + acc[t][r] + bias;
      }
    }
  }
}

// ---------------------------------------------------------------------------
extern "C" void kernel_launch(void* const* d_in, const int* in_sizes, int n_in,
                              void* d_out, int out_size, void* d_ws,
                              size_t ws_size, hipStream_t stream) {
  const float* hidden = (const float*)d_in[0];
  const float* wq = (const float*)d_in[1];
  const float* bq = (const float*)d_in[2];
  const float* wk = (const float*)d_in[3];
  const float* bk = (const float*)d_in[4];
  const float* wv = (const float*)d_in[5];
  const float* bv = (const float*)d_in[6];
  const float* wo = (const float*)d_in[7];
  const float* bo = (const float*)d_in[8];
  const float* ln_g = (const float*)d_in[9];
  const float* ln_b = (const float*)d_in[10];
  float* out = (float*)d_out;

  char* ws = (char*)d_ws;
  size_t off = 0;
  auto carve = [&](size_t bytes) -> void* {
    void* p = ws + off;
    off += (bytes + 255) & ~(size_t)255;
    return p;
  };
  _Float16* wqT    = (_Float16*)carve((size_t)HID * AHEAD * 2);
  _Float16* wkT    = (_Float16*)carve((size_t)HID * AHEAD * 2);
  _Float16* wvT    = (_Float16*)carve((size_t)HID * AHEAD * 2);
  _Float16* woT    = (_Float16*)carve((size_t)AHEAD * HID * 2);
  _Float16* normed = (_Float16*)carve((size_t)BATCH * NPAD * HID * 2);
  _Float16* Qb     = (_Float16*)carve((size_t)BATCH * NHEADS * NPAD * DHEAD * 2);
  _Float16* Kb     = (_Float16*)carve((size_t)BATCH * NHEADS * NPAD * DHEAD * 2);
  _Float16* Vt     = (_Float16*)carve((size_t)BATCH * NHEADS * DHEAD * NPAD * 2);
  _Float16* ctxb   = (_Float16*)carve((size_t)BATCH * NTOK * AHEAD * 2);

  const int wn = HID * AHEAD;
  k_transpose_f16<<<(wn + 255) / 256, 256, 0, stream>>>(wq, wqT, HID, AHEAD);
  k_transpose_f16<<<(wn + 255) / 256, 256, 0, stream>>>(wk, wkT, HID, AHEAD);
  k_transpose_f16<<<(wn + 255) / 256, 256, 0, stream>>>(wv, wvT, HID, AHEAD);
  k_transpose_f16<<<(wn + 255) / 256, 256, 0, stream>>>(wo, woT, AHEAD, HID);

  k_layernorm<<<BATCH * NPAD, 256, 0, stream>>>(hidden, ln_g, ln_b, normed);

  k_qkv<<<dim3(NPAD / 64, AHEAD / 64, BATCH * 3), 128, 0, stream>>>(
      normed, wqT, wkT, wvT, bq, bk, bv, Qb, Kb, Vt);

  k_attn<<<dim3(NPAD / 64, NHEADS, BATCH), 128, 0, stream>>>(Qb, Kb, Vt, ctxb);

  k_out<<<dim3((MROWS + 63) / 64, HID / 64), 128, 0, stream>>>(ctxb, woT, bo,
                                                              hidden, out);
}
